// SimpleLSTM_86277303042547
// MI455X (gfx1250) — compile-verified, matrix operand order checked by reading, not run on hardware
//
#include <hip/hip_runtime.h>
#include <cstdint>
#include <cstddef>

typedef __bf16 bf16;
typedef bf16  v16bf __attribute__((ext_vector_type(16)));
typedef float v8f   __attribute__((ext_vector_type(8)));
typedef unsigned int v4u __attribute__((ext_vector_type(4)));

#define BATCH 512
#define TLEN  2000
#define HDIM  128
#define GDIM  512   // 4*H
#define IN0   13
#define MTILE 16

// ---------------------------------------------------------------------------
// WMMA fragment helpers (CDNA5 16x16x32 bf16, wave32)
// A layout: lane m<16 -> row m, K {k0..k0+7, k0+16..k0+23}
//           lane m>=16 -> row m-16, K {k0+8..k0+15, k0+24..k0+31}
// B layout (col-striped mirror): lane n -> col n&15, same K split.
// ---------------------------------------------------------------------------
union Frag32 { v4u q[2]; v16bf v; };

// A-fragment load from an LDS tile (two 16B chunks, 32B apart)
__device__ __forceinline__ v16bf ld_frag(const bf16* base, int stride, int row, int kbase) {
  const bf16* p = base + row * stride + kbase;
  Frag32 f;
  f.q[0] = *(const v4u*)(p);        // K kbase .. kbase+7
  f.q[1] = *(const v4u*)(p + 16);   // K kbase+16 .. kbase+23
  return f.v;
}

// One-time B-fragment build straight from global f32 weights (per-lane slice).
__device__ __forceinline__ v16bf ld_wfrag(const float* __restrict__ W, int ldw,
                                          int row, int kbase) {
  Frag32 f;
  const float* p = W + (size_t)row * ldw + kbase;
#pragma unroll
  for (int i = 0; i < 8; ++i) {
    f.v[i]     = (bf16)p[i];
    f.v[i + 8] = (bf16)p[i + 16];
  }
  return f.v;
}

// Same, with zero padding beyond kmax (for the K=13 input projection).
__device__ __forceinline__ v16bf ld_wfrag_pad(const float* __restrict__ W, int ldw,
                                              int row, int kbase, int kmax) {
  Frag32 f;
#pragma unroll
  for (int i = 0; i < 8; ++i) {
    int k0 = kbase + i, k1 = kbase + 16 + i;
    f.v[i]     = (k0 < kmax) ? (bf16)W[(size_t)row * ldw + k0] : (bf16)0.0f;
    f.v[i + 8] = (k1 < kmax) ? (bf16)W[(size_t)row * ldw + k1] : (bf16)0.0f;
  }
  return f.v;
}

__device__ __forceinline__ v8f wmma_bf16(v16bf a, v16bf b, v8f c) {
  return __builtin_amdgcn_wmma_f32_16x16x32_bf16(false, a, false, b, (short)0, c, false, false);
}

__device__ __forceinline__ float sigmoidf(float x) { return 1.0f / (1.0f + __expf(-x)); }
__device__ __forceinline__ float tanh_f(float x)   { return 1.0f - 2.0f / (1.0f + __expf(2.0f * x)); }

// ---------------------------------------------------------------------------
// Layer 0: x[B,T,13] -> h1[T,B,H] (bf16).  32 WGs x 16 batch rows, 8 waves.
// All 20 B-fragments (W_ih padded + W_hh) live in registers for the whole scan.
// ---------------------------------------------------------------------------
__global__ __launch_bounds__(256) void lstm_layer0(
    const float* __restrict__ x,
    const float* __restrict__ Wih, const float* __restrict__ Whh,
    const float* __restrict__ bih, const float* __restrict__ bhh,
    bf16* __restrict__ h1out)
{
  __shared__ bf16 sH[2][MTILE][HDIM];       // 8 KB   (double-buffered h state tile)
  __shared__ bf16 sX[2][MTILE][32];         // 2 KB   (double-buffered x tile, K-padded)

  const int tid   = threadIdx.x;
  const int lane  = tid & 31;
  const int wv    = tid >> 5;        // 0..7
  const int lcol  = lane & 15;
  const int lhalf = lane >> 4;       // 0/1
  const int n0    = wv * 16;         // owned h-column base
  const int col   = n0 + lcol;       // owned h column (0..127)
  const int rowoff= lhalf * 8;
  const int b0    = blockIdx.x * MTILE;
  const int xr    = tid / IN0;       // valid for tid < 208
  const int xc    = tid - xr * IN0;

  // ---- loop-invariant B-fragments: registers for the whole 2000-step scan ----
  v16bf Bx[4], Bh[4][4];
#pragma unroll
  for (int q = 0; q < 4; ++q) {
    const int wrow = q * HDIM + col;
    Bx[q] = ld_wfrag_pad(Wih, IN0, wrow, lhalf * 8, IN0);
#pragma unroll
    for (int kt = 0; kt < 4; ++kt)
      Bh[q][kt] = ld_wfrag(Whh, HDIM, wrow, kt * 32 + lhalf * 8);
  }

  // zero state tiles + x pad columns
  {
    bf16* hz = &sH[0][0][0];
    for (int i = tid; i < 2 * MTILE * HDIM; i += 256) hz[i] = (bf16)0.0f;
    bf16* xz = &sX[0][0][0];
    for (int i = tid; i < 2 * MTILE * 32; i += 256) xz[i] = (bf16)0.0f;
  }
  __syncthreads();

  // stage x tile for t=0 into buffer 0
  if (tid < MTILE * IN0)
    sX[0][xr][xc] = (bf16)x[((size_t)(b0 + xr) * TLEN + 0) * IN0 + xc];

  float bias[4];
#pragma unroll
  for (int q = 0; q < 4; ++q) bias[q] = bih[q * HDIM + col] + bhh[q * HDIM + col];

  float cst[8];
#pragma unroll
  for (int r = 0; r < 8; ++r) cst[r] = 0.0f;

  __syncthreads();

  int p = 0;
  for (int t = 0; t < TLEN; ++t) {
    // A fragments: x tile (K=32 padded) and h state (K=128 -> 4 k-tiles)
    v16bf ax = ld_frag(&sX[p][0][0], 32, lcol, lhalf * 8);
    v16bf ah[4];
#pragma unroll
    for (int kt = 0; kt < 4; ++kt)
      ah[kt] = ld_frag(&sH[p][0][0], HDIM, lcol, kt * 32 + lhalf * 8);

    v8f acc[4];
#pragma unroll
    for (int q = 0; q < 4; ++q) {
#pragma unroll
      for (int r = 0; r < 8; ++r) acc[q][r] = bias[q];
    }

#pragma unroll
    for (int q = 0; q < 4; ++q) {
      acc[q] = wmma_bf16(ax, Bx[q], acc[q]);
#pragma unroll
      for (int kt = 0; kt < 4; ++kt)
        acc[q] = wmma_bf16(ah[kt], Bh[q][kt], acc[q]);
    }

    // gates -> new c,h (this lane owns col for rows rowoff..rowoff+7)
    bf16 hv[8];
#pragma unroll
    for (int r = 0; r < 8; ++r) {
      float gi = sigmoidf(acc[0][r]);
      float gf = sigmoidf(acc[1][r]);
      float gg = tanh_f(acc[2][r]);
      float go = sigmoidf(acc[3][r]);
      float c  = gf * cst[r] + gi * gg;
      cst[r]   = c;
      hv[r]    = (bf16)(go * tanh_f(c));
    }

    // stage next x tile into the other buffer
    if (t + 1 < TLEN && tid < MTILE * IN0)
      sX[1 - p][xr][xc] = (bf16)x[((size_t)(b0 + xr) * TLEN + (t + 1)) * IN0 + xc];

    // publish new h (LDS for next step's A, global [T,B,H] for layer 1)
#pragma unroll
    for (int r = 0; r < 8; ++r) {
      const int row = r + rowoff;
      sH[1 - p][row][col] = hv[r];
      h1out[((size_t)t * BATCH + b0 + row) * HDIM + col] = hv[r];
    }

    __syncthreads();
    p ^= 1;
  }
}

// ---------------------------------------------------------------------------
// Layer 1: h1[T,B,H] (bf16) -> h2 at t=T-1 (f32 [B,H]).
// All 32 B-fragments (W_ih + W_hh) register-resident (~256 VGPRs of weights;
// gfx1250 allows up to 1024 VGPRs/wave via VGPR_MSB).
// ---------------------------------------------------------------------------
__global__ __launch_bounds__(256) void lstm_layer1(
    const bf16* __restrict__ h1,
    const float* __restrict__ Wih, const float* __restrict__ Whh,
    const float* __restrict__ bih, const float* __restrict__ bhh,
    float* __restrict__ h2last)
{
  __shared__ bf16 sIn[2][MTILE][HDIM];     // 8 KB (double-buffered h1 input tile)
  __shared__ bf16 sH[2][MTILE][HDIM];      // 8 KB (double-buffered h2 state tile)

  const int tid   = threadIdx.x;
  const int lane  = tid & 31;
  const int wv    = tid >> 5;
  const int lcol  = lane & 15;
  const int lhalf = lane >> 4;
  const int n0    = wv * 16;
  const int col   = n0 + lcol;
  const int rowoff= lhalf * 8;
  const int b0    = blockIdx.x * MTILE;

  // loop-invariant B-fragments in registers
  v16bf Bi[4][4], Bh[4][4];
#pragma unroll
  for (int q = 0; q < 4; ++q) {
    const int wrow = q * HDIM + col;
#pragma unroll
    for (int kt = 0; kt < 4; ++kt) {
      Bi[q][kt] = ld_wfrag(Wih, HDIM, wrow, kt * 32 + lhalf * 8);
      Bh[q][kt] = ld_wfrag(Whh, HDIM, wrow, kt * 32 + lhalf * 8);
    }
  }

  {
    bf16* hz = &sH[0][0][0];
    for (int i = tid; i < 2 * MTILE * HDIM; i += 256) hz[i] = (bf16)0.0f;
  }
  __syncthreads();

  // stage h1 input tile for t=0 (coalesced: 256 threads x 16B = 4KB tile)
  ((v4u*)&sIn[0][0][0])[tid] =
      ((const v4u*)(h1 + ((size_t)0 * BATCH + b0) * HDIM))[tid];

  float bias[4];
#pragma unroll
  for (int q = 0; q < 4; ++q) bias[q] = bih[q * HDIM + col] + bhh[q * HDIM + col];

  float cst[8];
#pragma unroll
  for (int r = 0; r < 8; ++r) cst[r] = 0.0f;

  __syncthreads();

  int p = 0;
  for (int t = 0; t < TLEN; ++t) {
    v16bf ain[4], ah[4];
#pragma unroll
    for (int kt = 0; kt < 4; ++kt) {
      ain[kt] = ld_frag(&sIn[p][0][0], HDIM, lcol, kt * 32 + lhalf * 8);
      ah[kt]  = ld_frag(&sH[p][0][0],  HDIM, lcol, kt * 32 + lhalf * 8);
    }

    // prefetch next input tile to registers while WMMAs run
    v4u pf = {0u, 0u, 0u, 0u};
    if (t + 1 < TLEN)
      pf = ((const v4u*)(h1 + ((size_t)(t + 1) * BATCH + b0) * HDIM))[tid];

    v8f acc[4];
#pragma unroll
    for (int q = 0; q < 4; ++q) {
#pragma unroll
      for (int r = 0; r < 8; ++r) acc[q][r] = bias[q];
    }

#pragma unroll
    for (int q = 0; q < 4; ++q) {
#pragma unroll
      for (int kt = 0; kt < 4; ++kt)
        acc[q] = wmma_bf16(ain[kt], Bi[q][kt], acc[q]);
#pragma unroll
      for (int kt = 0; kt < 4; ++kt)
        acc[q] = wmma_bf16(ah[kt], Bh[q][kt], acc[q]);
    }

    float hf[8];
#pragma unroll
    for (int r = 0; r < 8; ++r) {
      float gi = sigmoidf(acc[0][r]);
      float gf = sigmoidf(acc[1][r]);
      float gg = tanh_f(acc[2][r]);
      float go = sigmoidf(acc[3][r]);
      float c  = gf * cst[r] + gi * gg;
      cst[r]   = c;
      hf[r]    = go * tanh_f(c);
    }

#pragma unroll
    for (int r = 0; r < 8; ++r)
      sH[1 - p][r + rowoff][col] = (bf16)hf[r];

    if (t + 1 < TLEN)
      ((v4u*)&sIn[1 - p][0][0])[tid] = pf;

    if (t == TLEN - 1) {
#pragma unroll
      for (int r = 0; r < 8; ++r)
        h2last[(size_t)(b0 + r + rowoff) * HDIM + col] = hf[r];
    }

    __syncthreads();
    p ^= 1;
  }
}

// ---------------------------------------------------------------------------
// FC head: out[B,4] = h2last[B,H] @ W_fc^T + b_fc   (trivial)
// ---------------------------------------------------------------------------
__global__ __launch_bounds__(256) void fc_head(
    const float* __restrict__ h2, const float* __restrict__ Wfc,
    const float* __restrict__ bfc, float* __restrict__ out)
{
  int b = blockIdx.x * blockDim.x + threadIdx.x;
  if (b >= BATCH) return;
  const float* hr = h2 + (size_t)b * HDIM;
#pragma unroll
  for (int c = 0; c < 4; ++c) {
    float s = bfc[c];
    const float* w = Wfc + c * HDIM;
    for (int k = 0; k < HDIM; ++k) s += hr[k] * w[k];
    out[b * 4 + c] = s;
  }
}

// ---------------------------------------------------------------------------
extern "C" void kernel_launch(void* const* d_in, const int* in_sizes, int n_in,
                              void* d_out, int out_size, void* d_ws, size_t ws_size,
                              hipStream_t stream) {
  (void)in_sizes; (void)n_in; (void)out_size; (void)ws_size;
  const float* x    = (const float*)d_in[0];
  const float* Wih0 = (const float*)d_in[1];
  const float* Whh0 = (const float*)d_in[2];
  const float* bih0 = (const float*)d_in[3];
  const float* bhh0 = (const float*)d_in[4];
  const float* Wih1 = (const float*)d_in[5];
  const float* Whh1 = (const float*)d_in[6];
  const float* bih1 = (const float*)d_in[7];
  const float* bhh1 = (const float*)d_in[8];
  const float* Wfc  = (const float*)d_in[9];
  const float* bfc  = (const float*)d_in[10];

  bf16*  h1     = (bf16*)d_ws;                                   // [T,B,H] bf16
  float* h2last = (float*)((char*)d_ws +
                           (size_t)TLEN * BATCH * HDIM * sizeof(bf16)); // [B,H] f32

  lstm_layer0<<<dim3(BATCH / MTILE), dim3(256), 0, stream>>>(
      x, Wih0, Whh0, bih0, bhh0, h1);
  lstm_layer1<<<dim3(BATCH / MTILE), dim3(256), 0, stream>>>(
      h1, Wih1, Whh1, bih1, bhh1, h2last);
  fc_head<<<dim3(2), dim3(256), 0, stream>>>(h2last, Wfc, bfc, (float*)d_out);
}
